// TRA32_83983790506209
// MI455X (gfx1250) — matrix operand order repaired and checked by reading, not compiled, not run on hardware
//
#include <hip/hip_runtime.h>

// ---------------------------------------------------------------------------
// Types
// ---------------------------------------------------------------------------
typedef __attribute__((ext_vector_type(16))) __bf16 v16bf;
typedef __attribute__((ext_vector_type(8)))  float  v8f;
typedef unsigned short u16;
typedef __attribute__((ext_vector_type(8))) u16 u16x8;
typedef __attribute__((ext_vector_type(4))) unsigned int uint32x4;
typedef __attribute__((ext_vector_type(8))) int int32x8;
typedef __attribute__((ext_vector_type(4))) int int32x4;

// 32-bit LDS (addrspace(3)) pointer to a 16-byte vector.
typedef __attribute__((address_space(3))) const u16x8* as3_v8p;

union Frag16 { u16x8 h[2]; v16bf v; };

#define BATCH 8
#define CH    64
#define NPIX  4096   // 64*64

// fp32 -> bf16, round-to-nearest-even
__device__ __forceinline__ u16 f2bf(float x) {
    unsigned u = __float_as_uint(x);
    u += 0x7fffu + ((u >> 16) & 1u);
    return (u16)(u >> 16);
}

__device__ __forceinline__ v8f wmma_bf16(v16bf a, v16bf b, v8f c) {
    return __builtin_amdgcn_wmma_f32_16x16x32_bf16(
        false, a, false, b, (short)0, c, false, false);
}

// Low 32 bits of a generic pointer to LDS == LDS byte address (ISA aperture
// mapping: the LDS offset is the pointer's low 32 bits).
__device__ __forceinline__ unsigned ldsoff(const void* p) {
    return (unsigned)(uintptr_t)p;
}

// A-matrix fragment (16x32 bf16) from row-major (rows x ldc) global buffer.
__device__ __forceinline__ v16bf load_frag_A(const u16* __restrict__ base,
                                             int row0, int ldc, int k0, int lane) {
    int r  = lane & 15;
    int hh = lane >> 4;
    const u16* p = base + (size_t)(row0 + r) * ldc + k0;
    Frag16 f;
    f.h[0] = *(const u16x8*)(p + hh * 8);
    f.h[1] = *(const u16x8*)(p + 16 + hh * 8);
    return f.v;
}

// B-matrix fragment (32x16 bf16) from an LDS tile stored row-major
// [cols][rowlen]: column n contiguous over K.
// The tiles are written by the Tensor Data Mover, invisible to the compiler.
// Launder the LDS byte offset through an empty asm so the compiler cannot
// prove the memory unwritten (else loads fold to undef) while keeping plain
// batched ds_load_b128s (no volatile -> no flat/SYS-scope serialization).
__device__ __forceinline__ v16bf frag_B_lds(const u16* tile, int rowlen,
                                            int col0, int k0, int lane) {
    int n  = lane & 15;
    int hh = lane >> 4;
    unsigned off = ldsoff(tile + (col0 + n) * rowlen + k0 + hh * 8);
    asm volatile("" : "+v"(off));        // hide provenance
    as3_v8p p = (as3_v8p)off;
    Frag16 f;
    f.h[0] = p[0];
    f.h[1] = p[2];
    return f.v;
}

// Wave-uniform "am I wave 0" as an SGPR condition: guarantees a scalar
// branch (s_cbranch), never exec-predication. TDM ops ignore EXEC, so they
// must be branched around, not predicated.
__device__ __forceinline__ bool is_wave0() {
    return __builtin_amdgcn_readfirstlane(threadIdx.x >> 5) == 0;
}

// ---------------------------------------------------------------------------
// Tensor Data Mover: 2-D tile (tile_d0 contiguous elems x tile_d1 rows,
// row stride = stride_elems) of bf16 from global -> LDS.
// D# fields per CDNA5 ISA 8.3/8.4; groups 2/3 zero (2-D tensor).
// ---------------------------------------------------------------------------
__device__ __forceinline__ void tdm_load_2d(unsigned lds_byte_addr,
                                            const u16* gptr,
                                            unsigned tile_d0, unsigned tile_d1,
                                            unsigned long long stride_elems) {
    unsigned long long ga = (unsigned long long)(uintptr_t)gptr;
    uint32x4 g0;
    g0[0] = 1u;                                   // count=1, user descriptor
    g0[1] = lds_byte_addr;                        // lds_addr
    g0[2] = (unsigned)(ga & 0xffffffffu);         // global_addr[31:0]
    g0[3] = (unsigned)((ga >> 32) & 0x01ffffffu)  // global_addr[56:32]
          | (2u << 30);                           // type=2 ("image")
    int32x8 g1;
    g1[0] = (int)(1u << 16);                      // wg_mask=0, data_size=1 (2B)
    g1[1] = (int)((tile_d0 & 0xffffu) << 16);     // tensor_dim0[15:0]
    g1[2] = (int)(((tile_d0 >> 16) & 0xffffu) | ((tile_d1 & 0xffffu) << 16)); // dim0 hi, dim1 lo
    g1[3] = (int)(((tile_d1 >> 16) & 0xffffu) | ((tile_d0 & 0xffffu) << 16)); // dim1 hi, tile_dim0
    g1[4] = (int)(tile_d1 & 0xffffu);             // tile_dim1 (tile_dim2=0)
    g1[5] = (int)(unsigned)(stride_elems & 0xffffffffu);      // dim0_stride[31:0]
    g1[6] = (int)(unsigned)((stride_elems >> 32) & 0xffffu);  // dim0_stride[47:32]
    g1[7] = 0;
    int32x4 z4 = {0, 0, 0, 0};
    int32x8 z8 = {0, 0, 0, 0, 0, 0, 0, 0};
    __builtin_amdgcn_tensor_load_to_lds(g0, g1, z4, z4, z8, 0);
}

// ---------------------------------------------------------------------------
// Kernel 1: q/k/v projections.
//   qT, kT : (B, N, C) bf16 ; vB : (B, C, N) bf16
// ---------------------------------------------------------------------------
__global__ __launch_bounds__(256)
void qkv_kernel(const float* __restrict__ x,
                const float* __restrict__ Wq, const float* __restrict__ bq,
                const float* __restrict__ Wk, const float* __restrict__ bk,
                const float* __restrict__ Wv, const float* __restrict__ bv,
                u16* __restrict__ qT, u16* __restrict__ kT, u16* __restrict__ vB) {
    int p = blockIdx.x * blockDim.x + threadIdx.x;
    int b = p >> 12;
    int i = p & (NPIX - 1);

    const float* xp = x + (size_t)b * CH * NPIX + i;
    float xr[CH];
#pragma unroll
    for (int c = 0; c < CH; ++c) xr[c] = xp[(size_t)c * NPIX];

    u16* qrow = qT + ((size_t)b * NPIX + i) * CH;
    u16* krow = kT + ((size_t)b * NPIX + i) * CH;
    u16* vcol = vB + (size_t)b * CH * NPIX + i;

    for (int o = 0; o < CH; ++o) {
        float aq = bq[o], ak = bk[o], av = bv[o];
#pragma unroll
        for (int c = 0; c < CH; ++c) {
            aq = fmaf(Wq[o * CH + c], xr[c], aq);
            ak = fmaf(Wk[o * CH + c], xr[c], ak);
            av = fmaf(Wv[o * CH + c], xr[c], av);
        }
        qrow[o] = f2bf(aq);
        krow[o] = f2bf(ak);
        vcol[(size_t)o * NPIX] = f2bf(av);
    }
}

// ---------------------------------------------------------------------------
// Kernel 2 (pass A): per-row stats of scores[j,i] = q_j . k_i.
// k tiles staged via TDM, triple-buffered. All four B-fragments are loaded
// into distinct values before the WMMA chain so the 8 ds_load_b128s issue
// back-to-back and the dscnt waits stagger under the matrix pipe.
// Lane-private online softmax; single cross-lane (m,Z) merge at the end.
// ---------------------------------------------------------------------------
__global__ __launch_bounds__(128)
void rowstats_kernel(const u16* __restrict__ qT, const u16* __restrict__ kT,
                     float* __restrict__ mout, float* __restrict__ rzout) {
    __shared__ __align__(16) u16 smem[3 * 32 * 64];   // 3 x 4KB k tiles

    int b    = blockIdx.x >> 6;
    int jblk = blockIdx.x & 63;
    int wave = threadIdx.x >> 5;
    int lane = threadIdx.x & 31;
    int j0   = jblk * 64 + wave * 16;

    const u16* qb = qT + (size_t)b * NPIX * CH;
    const u16* kb = kT + (size_t)b * NPIX * CH;

    v16bf a0 = load_frag_A(qb, j0, CH, 0, lane);
    v16bf a1 = load_frag_A(qb, j0, CH, 32, lane);

    float m[8], Z[8];
#pragma unroll
    for (int g = 0; g < 8; ++g) { m[g] = -1e30f; Z[g] = 0.0f; }

    // Prologue: DMA tile for step 0 into buffer 0.
    if (is_wave0())
        tdm_load_2d(ldsoff(smem), kb, 64, 32, 64);

    int cur = 0, nxt = 1;
    for (int s = 0; s < NPIX / 32; ++s) {
        int jn = (s + 1 < NPIX / 32) ? (s + 1) * 32 : 0;
        if (is_wave0())
            tdm_load_2d(ldsoff(smem + nxt * 32 * 64), kb + (size_t)jn * CH, 64, 32, 64);
        __builtin_amdgcn_s_wait_tensorcnt(1);   // previous tile landed
        __syncthreads();

        const u16* kc = smem + cur * (32 * 64);

        // Issue all fragment loads first (distinct destinations).
        v16bf b00 = frag_B_lds(kc, 64, 0, 0, lane);
        v16bf b01 = frag_B_lds(kc, 64, 0, 32, lane);
        v16bf b10 = frag_B_lds(kc, 64, 16, 0, lane);
        v16bf b11 = frag_B_lds(kc, 64, 16, 32, lane);

        v8f s0 = {}, s1 = {};
        s0 = wmma_bf16(a0, b00, s0);
        s0 = wmma_bf16(a1, b01, s0);
        s1 = wmma_bf16(a0, b10, s1);
        s1 = wmma_bf16(a1, b11, s1);

        // Lane-private online softmax over this lane's columns.
#pragma unroll
        for (int g = 0; g < 8; ++g) {
            float t  = fmaxf(s0[g], s1[g]);
            float mn = fmaxf(m[g], t);
            Z[g] = Z[g] * __expf(m[g] - mn)
                 + __expf(s0[g] - mn) + __expf(s1[g] - mn);
            m[g] = mn;
        }
        cur = nxt;
        nxt = (nxt == 2) ? 0 : nxt + 1;
    }

    // Cross-lane merge of (m, Z) within each 16-lane half (once).
#pragma unroll
    for (int g = 0; g < 8; ++g) {
#pragma unroll
        for (int mask = 1; mask <= 8; mask <<= 1) {
            float mo = __shfl_xor(m[g], mask, 32);
            float Zo = __shfl_xor(Z[g], mask, 32);
            float mn = fmaxf(m[g], mo);
            Z[g] = Z[g] * __expf(m[g] - mn) + Zo * __expf(mo - mn);
            m[g] = mn;
        }
    }

    if ((lane & 15) == 0) {
        int hh = lane >> 4;
#pragma unroll
        for (int g = 0; g < 8; ++g) {
            int j = j0 + g + 8 * hh;
            mout[(size_t)b * NPIX + j]  = m[g];
            rzout[(size_t)b * NPIX + j] = 1.0f / Z[g];
        }
    }
}

// ---------------------------------------------------------------------------
// Kernel 3 (pass B): res[c,i] = sum_j v[c,j] * exp(s[j,i]-m_j)*rZ_j.
// S'[i,j] = k_i . q_j (A = kT rows, B = q cols). q and v tiles staged via
// TDM (shared by all 4 waves), triple-buffered. q/v fragments and the m/rZ
// scalars are all loaded up front so their latencies hide under the WMMA +
// exp chain. Column-normalized P goes D-layout -> A-layout through per-wave
// LDS staging, then O += P x V^T.
// ---------------------------------------------------------------------------
__global__ __launch_bounds__(128)
void attnpv_kernel(const u16* __restrict__ qT, const u16* __restrict__ kT,
                   const u16* __restrict__ vB,
                   const float* __restrict__ mbuf, const float* __restrict__ rzbuf,
                   float* __restrict__ resT) {
    // [0, 12KB)   : 3 x q tile (32 j rows x 64 c)
    // [12KB,24KB) : 3 x v tile (64 c rows x 32 j)
    // [24KB,28KB) : 4 x per-wave P staging (16 x 32)
    __shared__ __align__(16) u16 smem[3 * 2048 + 3 * 2048 + 4 * 512];
    u16* qtile = smem;
    u16* vtile = smem + 3 * 2048;
    u16* Ptile = smem + 6 * 2048;

    int b    = blockIdx.x >> 6;
    int iblk = blockIdx.x & 63;
    int wave = threadIdx.x >> 5;
    int lane = threadIdx.x & 31;
    int i0   = iblk * 64 + wave * 16;
    int r    = lane & 15;
    int hh   = lane >> 4;

    const u16* qb = qT + (size_t)b * NPIX * CH;
    const u16* kb = kT + (size_t)b * NPIX * CH;
    const u16* vb = vB + (size_t)b * CH * NPIX;
    const float* mb = mbuf  + (size_t)b * NPIX;
    const float* rb = rzbuf + (size_t)b * NPIX;

    v16bf a0 = load_frag_A(kb, i0, CH, 0, lane);
    v16bf a1 = load_frag_A(kb, i0, CH, 32, lane);

    v8f O[4];
#pragma unroll
    for (int t = 0; t < 4; ++t) O[t] = (v8f){};

    // Prologue: DMA q/v tiles for step 0 into buffer 0.
    if (is_wave0()) {
        tdm_load_2d(ldsoff(qtile), qb, 64, 32, 64);      // 32 rows x 64, stride 64
        tdm_load_2d(ldsoff(vtile), vb, 32, 64, NPIX);    // 64 rows x 32, stride 4096
    }

    int cur = 0, nxt = 1;
    for (int s = 0; s < NPIX / 32; ++s) {
        int j0 = s * 32;
        int jn = (s + 1 < NPIX / 32) ? (s + 1) * 32 : 0;
        if (is_wave0()) {
            tdm_load_2d(ldsoff(qtile + nxt * 2048), qb + (size_t)jn * CH, 64, 32, 64);
            tdm_load_2d(ldsoff(vtile + nxt * 2048), vb + jn, 32, 64, NPIX);
        }
        __builtin_amdgcn_s_wait_tensorcnt(2);   // previous pair landed
        __syncthreads();

        const u16* qc = qtile + cur * 2048;
        const u16* vc = vtile + cur * 2048;

        // Per-column (j) normalization constants — issue early (loadcnt
        // latency hides under the score WMMAs).
        float m0 = mb[j0 + r],      z0 = rb[j0 + r];
        float m1 = mb[j0 + 16 + r], z1 = rb[j0 + 16 + r];

        // Issue all q fragment loads first (distinct destinations).
        v16bf q00 = frag_B_lds(qc, 64, 0, 0, lane);
        v16bf q01 = frag_B_lds(qc, 64, 0, 32, lane);
        v16bf q10 = frag_B_lds(qc, 64, 16, 0, lane);
        v16bf q11 = frag_B_lds(qc, 64, 16, 32, lane);

        // v fragments too — their DS latency hides under WMMA + exp work.
        v16bf vf0 = frag_B_lds(vc, 32, 0, 0, lane);
        v16bf vf1 = frag_B_lds(vc, 32, 16, 0, lane);
        v16bf vf2 = frag_B_lds(vc, 32, 32, 0, lane);
        v16bf vf3 = frag_B_lds(vc, 32, 48, 0, lane);

        v8f s0 = {}, s1 = {};
        s0 = wmma_bf16(a0, q00, s0);
        s0 = wmma_bf16(a1, q01, s0);
        s1 = wmma_bf16(a0, q10, s1);
        s1 = wmma_bf16(a1, q11, s1);

        // D-layout -> LDS (row-major 16x32 P tile, per wave).
        u16* Pw = Ptile + wave * 512;
#pragma unroll
        for (int g = 0; g < 8; ++g) {
            int row = g + 8 * hh;
            Pw[row * 32 + r]      = f2bf(__expf(s0[g] - m0) * z0);
            Pw[row * 32 + 16 + r] = f2bf(__expf(s1[g] - m1) * z1);
        }

        // Re-load P as A fragment; within-wave LDS ops are in-order.
        Frag16 pf;
        pf.h[0] = *(const u16x8*)&Pw[r * 32 + hh * 8];
        pf.h[1] = *(const u16x8*)&Pw[r * 32 + 16 + hh * 8];

        O[0] = wmma_bf16(pf.v, vf0, O[0]);
        O[1] = wmma_bf16(pf.v, vf1, O[1]);
        O[2] = wmma_bf16(pf.v, vf2, O[2]);
        O[3] = wmma_bf16(pf.v, vf3, O[3]);

        cur = nxt;
        nxt = (nxt == 2) ? 0 : nxt + 1;
    }

    // Store O: element (i = i0 + g + 8*hh, c = t*16 + r), resT is (B,N,C) f32.
#pragma unroll
    for (int t = 0; t < 4; ++t)
#pragma unroll
        for (int g = 0; g < 8; ++g)
            resT[((size_t)b * NPIX + i0 + g + 8 * hh) * CH + t * 16 + r] = O[t][g];
}

// ---------------------------------------------------------------------------
// Kernel 4: out = Wo*res + bo + res, output (B, C, N) f32.
// ---------------------------------------------------------------------------
__global__ __launch_bounds__(256)
void outproj_kernel(const float* __restrict__ resT,
                    const float* __restrict__ Wo, const float* __restrict__ bo,
                    float* __restrict__ out) {
    int p = blockIdx.x * blockDim.x + threadIdx.x;
    int b = p >> 12;
    int i = p & (NPIX - 1);

    const float* rp = resT + ((size_t)b * NPIX + i) * CH;
    float rr[CH];
#pragma unroll
    for (int c = 0; c < CH; ++c) rr[c] = rp[c];

    float* op = out + (size_t)b * CH * NPIX + i;
    for (int o = 0; o < CH; ++o) {
        float acc = bo[o] + rr[o];
#pragma unroll
        for (int c = 0; c < CH; ++c) acc = fmaf(Wo[o * CH + c], rr[c], acc);
        op[(size_t)o * NPIX] = acc;
    }
}

// ---------------------------------------------------------------------------
// Launch
// ---------------------------------------------------------------------------
extern "C" void kernel_launch(void* const* d_in, const int* in_sizes, int n_in,
                              void* d_out, int out_size, void* d_ws, size_t ws_size,
                              hipStream_t stream) {
    const float* x  = (const float*)d_in[0];
    const float* Wq = (const float*)d_in[1];
    const float* bq = (const float*)d_in[2];
    const float* Wk = (const float*)d_in[3];
    const float* bk = (const float*)d_in[4];
    const float* Wv = (const float*)d_in[5];
    const float* bv = (const float*)d_in[6];
    const float* Wo = (const float*)d_in[7];
    const float* bo = (const float*)d_in[8];
    float* out = (float*)d_out;

    char* ws = (char*)d_ws;
    u16*   qT   = (u16*)(ws + (size_t)0);                 // 4 MiB
    u16*   kT   = (u16*)(ws + ((size_t)4 << 20));         // 4 MiB
    u16*   vB   = (u16*)(ws + ((size_t)8 << 20));         // 4 MiB
    float* mbuf = (float*)(ws + ((size_t)12 << 20));      // 128 KiB
    float* rz   = (float*)(ws + ((size_t)12 << 20) + (1 << 18));
    float* resT = (float*)(ws + ((size_t)13 << 20));      // 8 MiB

    qkv_kernel<<<dim3((BATCH * NPIX) / 256), dim3(256), 0, stream>>>(
        x, Wq, bq, Wk, bk, Wv, bv, qT, kT, vB);

    rowstats_kernel<<<dim3(BATCH * (NPIX / 64)), dim3(128), 0, stream>>>(
        qT, kT, mbuf, rz);

    attnpv_kernel<<<dim3(BATCH * (NPIX / 64)), dim3(128), 0, stream>>>(
        qT, kT, vB, mbuf, rz, resT);

    outproj_kernel<<<dim3((BATCH * NPIX) / 256), dim3(256), 0, stream>>>(
        resT, Wo, bo, out);
}
